// ChildSumTreeLSTM_50912542327444
// MI455X (gfx1250) — matrix operand order repaired and checked by reading, not compiled, not run on hardware
//
#include <hip/hip_runtime.h>
#include <hip/hip_bf16.h>
#include <math.h>

// ---------------------------------------------------------------------------
// Child-Sum TreeLSTM on MI455X (gfx1250, wave32, WMMA).
//
// Per level l (m = 2^l nodes), two bf16 WMMA GEMMs:
//   node_level: pre_* = x @ W_*^T  (fused epilogue: + child g_* gathers,
//               sigmoid/tanh, c/h update)
//   gu_level:   g_*   = h @ U_*^T  (per-node, consumed by parent level)
// Weights packed once into WMMA B-fragment layout (bf16).
// Workspace layout (bytes):            requires ~145 MiB of d_ws
//   [0,16Mi)    WPACK  bf16 packed W_i,W_o,W_u,W_f,U_i,U_o,U_u,U_f
//   [16,80Mi)   GU     bf16 [8192][4][1024]
//   [80,112Mi)  CA     f32  [8192][1024]
//   [112,128Mi) CB     f32  [4096][1024]
//   [128,144Mi) HB     bf16 [8192][1024]
//   [144Mi,..)  INITV  f32  5*1024 (h_init @ U contributions for leaf level)
// ---------------------------------------------------------------------------

typedef __bf16 v16bf __attribute__((ext_vector_type(16)));
typedef __bf16 v8bf  __attribute__((ext_vector_type(8)));
typedef float  v8f   __attribute__((ext_vector_type(8)));

__device__ __forceinline__ unsigned short f2bf(float f) {
    unsigned int u = __float_as_uint(f);
    u += 0x7fffu + ((u >> 16) & 1u);           // round-to-nearest-even
    return (unsigned short)(u >> 16);
}
__device__ __forceinline__ float bf2f(unsigned short u) {
    return __uint_as_float(((unsigned int)u) << 16);
}
__device__ __forceinline__ float sigmoidf_(float z) {
    return 1.0f / (1.0f + __expf(-z));
}

// --- pack fp32 [1024x1024] row-major (n,k) weights into WMMA B-fragment bf16.
// Element layout: idx = (((mat*64+nt)*32+kt)*32+lane)*16 + e
//   n = nt*16 + (lane&15);  k = kt*32 + (lane<16?0:8) + (e&7) + (e<8?0:16)
__global__ __launch_bounds__(256) void pack_weights_k(
    const float* __restrict__ w0, const float* __restrict__ w1,
    const float* __restrict__ w2, const float* __restrict__ w3,
    const float* __restrict__ w4, const float* __restrict__ w5,
    const float* __restrict__ w6, const float* __restrict__ w7,
    unsigned short* __restrict__ wpack)
{
    int gid = blockIdx.x * 256 + threadIdx.x;          // 8*1024*1024 total
    int e    = gid & 15;
    int lane = (gid >> 4) & 31;
    int kt   = (gid >> 9) & 31;
    int nt   = (gid >> 14) & 63;
    int mat  = (gid >> 20) & 7;
    const float* W;
    switch (mat) {
        case 0: W = w0; break; case 1: W = w1; break;
        case 2: W = w2; break; case 3: W = w3; break;
        case 4: W = w4; break; case 5: W = w5; break;
        case 6: W = w6; break; default: W = w7; break;
    }
    int n = nt * 16 + (lane & 15);
    int k = kt * 32 + ((lane < 16) ? 0 : 8) + (e & 7) + ((e < 8) ? 0 : 16);
    wpack[gid] = f2bf(W[(size_t)n * 1024 + k]);
}

// --- precompute leaf-level U contributions from broadcast h_init.
// initv: [0,1024)=hsum@U_i, [1024)=hsum@U_o, [2048)=hsum@U_u,
//        [3072)=h0@U_f, [4096)=h1@U_f
__global__ __launch_bounds__(256) void init_vectors_k(
    const float* __restrict__ Ui, const float* __restrict__ Uo,
    const float* __restrict__ Uu, const float* __restrict__ Uf,
    const float* __restrict__ h_init, float* __restrict__ initv)
{
    int n = blockIdx.x * 256 + threadIdx.x;
    if (n >= 1024) return;
    float si = 0.f, so = 0.f, su = 0.f, sf0 = 0.f, sf1 = 0.f;
    for (int k = 0; k < 1024; ++k) {
        float h0 = h_init[k], h1 = h_init[1024 + k];
        float hs = h0 + h1;
        si  += hs * Ui[(size_t)n * 1024 + k];
        so  += hs * Uo[(size_t)n * 1024 + k];
        su  += hs * Uu[(size_t)n * 1024 + k];
        sf0 += h0 * Uf[(size_t)n * 1024 + k];
        sf1 += h1 * Uf[(size_t)n * 1024 + k];
    }
    initv[n] = si; initv[1024 + n] = so; initv[2048 + n] = su;
    initv[3072 + n] = sf0; initv[4096 + n] = sf1;
}

// --- fused node step: pre = x @ W_*^T (WMMA), then gate math + c/h update.
// mode 0: leaves (x = tokens[leaf_ids], child state = init vectors)
// mode 1: inner  (x = op_emb[op_ids[off+j]], children = 2j, 2j+1)
__global__ __launch_bounds__(128) void node_level_k(
    int mode, int m,
    const float* __restrict__ tokens, const int* __restrict__ leaf_ids,
    const float* __restrict__ op_emb, const int* __restrict__ op_ids, int op_off,
    const unsigned short* __restrict__ wpack,
    const float* __restrict__ c_child,          // mode 1: [2m][1024]
    const unsigned short* __restrict__ gu,      // mode 1: [2m][4][1024] bf16
    const float* __restrict__ initv,            // mode 0
    const float* __restrict__ c_init,           // [2][1024]
    const float* __restrict__ bi_p, const float* __restrict__ bo_p,
    const float* __restrict__ bu_p, const float* __restrict__ bf_p,
    float* __restrict__ c_out, unsigned short* __restrict__ h_out,
    int is_root, float* __restrict__ final_out)
{
    __shared__ unsigned short xs[16 * 1024];    // 16 rows x 1024 K, bf16 bits

    const int t  = threadIdx.x;
    const int rt = blockIdx.x;                  // 16-row tile of parents
    const int ct = blockIdx.y;                  // 64-col super tile

    // ---- gather + fp32->bf16 convert x tile into LDS (row-major) ----
    for (int idx = t; idx < 16 * 1024; idx += 128) {
        int r = idx >> 10, k = idx & 1023;
        int j = rt * 16 + r; if (j >= m) j = m - 1;     // clamp, rows unused
        const float* src = (mode == 0)
            ? tokens + (size_t)leaf_ids[j] * 1024
            : op_emb + (size_t)op_ids[op_off + j] * 1024;
        xs[idx] = f2bf(src[k]);
    }
    __syncthreads();

    const int w     = t >> 5;
    const int lane  = t & 31;
    const int lhalf = (lane < 16) ? 0 : 1;
    const int lmod  = lane & 15;
    const int nt    = ct * 4 + w;               // 16-col tile index in [0,64)

    v8f acc[4];
    #pragma unroll
    for (int g = 0; g < 4; ++g) acc[g] = v8f{0.f,0.f,0.f,0.f,0.f,0.f,0.f,0.f};

    const unsigned short* bptr[4];
    #pragma unroll
    for (int g = 0; g < 4; ++g)
        bptr[g] = wpack + ((size_t)((g * 64 + nt) * 32) * 32 + lane) * 16;

    // A fragment: lane holds row M=lmod, K = kk*32 + lhalf*8 + {0..7, 16..23}
    const unsigned short* arow = xs + lmod * 1024 + lhalf * 8;

    for (int kk = 0; kk < 32; ++kk) {
        union { v16bf v; struct { v8bf lo, hi; } s; } a;
        a.s.lo = *reinterpret_cast<const v8bf*>(arow + kk * 32);
        a.s.hi = *reinterpret_cast<const v8bf*>(arow + kk * 32 + 16);
        #pragma unroll
        for (int g = 0; g < 4; ++g) {
            v16bf b = *reinterpret_cast<const v16bf*>(bptr[g] + kk * 512);
            acc[g] = __builtin_amdgcn_wmma_f32_16x16x32_bf16(
                false, a.v, false, b, (short)0, acc[g], false, false);
        }
    }

    // ---- epilogue: C/D layout lane->(M = v + lhalf*8, N = lmod) ----
    const int col = ct * 64 + w * 16 + lmod;
    const float bi_ = bi_p[col], bo_ = bo_p[col], bu_ = bu_p[col], bf_ = bf_p[col];

    float gi0 = 0.f, go0 = 0.f, gu0 = 0.f, gf0v = 0.f, gf1v = 0.f, c0v = 0.f, c1v = 0.f;
    if (mode == 0) {
        gi0  = initv[col];        go0 = initv[1024 + col];
        gu0  = initv[2048 + col]; gf0v = initv[3072 + col]; gf1v = initv[4096 + col];
        c0v  = c_init[col];       c1v = c_init[1024 + col];
    }

    #pragma unroll
    for (int v = 0; v < 8; ++v) {
        int M = v + lhalf * 8;
        int j = rt * 16 + M;
        if (j >= m) continue;
        float pi = acc[0][v], po = acc[1][v], pu = acc[2][v], pf = acc[3][v];
        float gi, go, gg, gfa, gfb, c0, c1;
        if (mode == 0) {
            gi = gi0; go = go0; gg = gu0; gfa = gf0v; gfb = gf1v; c0 = c0v; c1 = c1v;
        } else {
            size_t ca_ = (size_t)(2 * j), cb_ = ca_ + 1;
            const unsigned short* g0 = gu + ca_ * 4096;
            const unsigned short* g1 = gu + cb_ * 4096;
            gi  = bf2f(g0[col])        + bf2f(g1[col]);
            go  = bf2f(g0[1024 + col]) + bf2f(g1[1024 + col]);
            gg  = bf2f(g0[2048 + col]) + bf2f(g1[2048 + col]);
            gfa = bf2f(g0[3072 + col]);
            gfb = bf2f(g1[3072 + col]);
            c0 = c_child[ca_ * 1024 + col];
            c1 = c_child[cb_ * 1024 + col];
        }
        float i_ = sigmoidf_(pi + gi + bi_);
        float o_ = sigmoidf_(po + go + bo_);
        float u_ = tanhf(pu + gg + bu_);
        float f0 = sigmoidf_(pf + gfa + bf_);
        float f1 = sigmoidf_(pf + gfb + bf_);
        float c  = i_ * u_ + f0 * c0 + f1 * c1;
        float h  = o_ * tanhf(c);
        c_out[(size_t)j * 1024 + col] = c;
        h_out[(size_t)j * 1024 + col] = f2bf(h);
        if (is_root && j == 0) {
            final_out[col]        = c;
            final_out[1024 + col] = h;
        }
    }
}

// --- g_* = h @ U_*^T for the m just-computed nodes (consumed by parents).
__global__ __launch_bounds__(128) void gu_level_k(
    int m, const unsigned short* __restrict__ hb,
    const unsigned short* __restrict__ wpack,   // U_* are matrices 4..7
    unsigned short* __restrict__ gu_out)
{
    __shared__ unsigned short hs[16 * 1024];

    const int t  = threadIdx.x;
    const int rt = blockIdx.x;
    const int ct = blockIdx.y;

    for (int idx = t; idx < 16 * 1024; idx += 128) {
        int r = idx >> 10, k = idx & 1023;
        int j = rt * 16 + r; if (j >= m) j = m - 1;
        hs[idx] = hb[(size_t)j * 1024 + k];
    }
    __syncthreads();

    const int w = t >> 5, lane = t & 31;
    const int lhalf = (lane < 16) ? 0 : 1;
    const int lmod  = lane & 15;
    const int nt    = ct * 4 + w;

    v8f acc[4];
    #pragma unroll
    for (int g = 0; g < 4; ++g) acc[g] = v8f{0.f,0.f,0.f,0.f,0.f,0.f,0.f,0.f};

    const unsigned short* bptr[4];
    #pragma unroll
    for (int g = 0; g < 4; ++g)
        bptr[g] = wpack + ((size_t)(((g + 4) * 64 + nt) * 32) * 32 + lane) * 16;

    const unsigned short* arow = hs + lmod * 1024 + lhalf * 8;

    for (int kk = 0; kk < 32; ++kk) {
        union { v16bf v; struct { v8bf lo, hi; } s; } a;
        a.s.lo = *reinterpret_cast<const v8bf*>(arow + kk * 32);
        a.s.hi = *reinterpret_cast<const v8bf*>(arow + kk * 32 + 16);
        #pragma unroll
        for (int g = 0; g < 4; ++g) {
            v16bf b = *reinterpret_cast<const v16bf*>(bptr[g] + kk * 512);
            acc[g] = __builtin_amdgcn_wmma_f32_16x16x32_bf16(
                false, a.v, false, b, (short)0, acc[g], false, false);
        }
    }

    const int col = ct * 64 + w * 16 + lmod;
    #pragma unroll
    for (int v = 0; v < 8; ++v) {
        int M = v + lhalf * 8;
        int j = rt * 16 + M;
        if (j >= m) continue;
        unsigned short* dst = gu_out + (size_t)j * 4096;
        #pragma unroll
        for (int g = 0; g < 4; ++g)
            dst[g * 1024 + col] = f2bf(acc[g][v]);
    }
}

extern "C" void kernel_launch(void* const* d_in, const int* in_sizes, int n_in,
                              void* d_out, int out_size, void* d_ws, size_t ws_size,
                              hipStream_t stream) {
    (void)in_sizes; (void)n_in; (void)out_size; (void)ws_size;

    const float* tokens   = (const float*)d_in[0];
    const int*   leaf_ids = (const int*)d_in[1];
    const int*   op_ids   = (const int*)d_in[2];
    const float* W_i = (const float*)d_in[3];
    const float* W_o = (const float*)d_in[4];
    const float* W_u = (const float*)d_in[5];
    const float* W_f = (const float*)d_in[6];
    const float* U_i = (const float*)d_in[7];
    const float* U_o = (const float*)d_in[8];
    const float* U_u = (const float*)d_in[9];
    const float* U_f = (const float*)d_in[10];
    const float* b_i = (const float*)d_in[11];
    const float* b_o = (const float*)d_in[12];
    const float* b_u = (const float*)d_in[13];
    const float* b_f = (const float*)d_in[14];
    const float* op_emb = (const float*)d_in[15];
    const float* c_init = (const float*)d_in[16];
    const float* h_init = (const float*)d_in[17];

    char* ws = (char*)d_ws;
    unsigned short* WPACK = (unsigned short*)(ws);
    unsigned short* GU    = (unsigned short*)(ws + ((size_t)16  << 20));
    float*          CA    = (float*)         (ws + ((size_t)80  << 20));
    float*          CB    = (float*)         (ws + ((size_t)112 << 20));
    unsigned short* HB    = (unsigned short*)(ws + ((size_t)128 << 20));
    float*          INITV = (float*)         (ws + ((size_t)144 << 20));

    // one-time weight packing + leaf U-contribution vectors
    pack_weights_k<<<32768, 256, 0, stream>>>(W_i, W_o, W_u, W_f,
                                              U_i, U_o, U_u, U_f, WPACK);
    init_vectors_k<<<4, 256, 0, stream>>>(U_i, U_o, U_u, U_f, h_init, INITV);

    // leaves (level 13): m = 8192
    node_level_k<<<dim3(512, 16), 128, 0, stream>>>(
        0, 8192, tokens, leaf_ids, nullptr, nullptr, 0,
        WPACK, nullptr, nullptr, INITV, c_init,
        b_i, b_o, b_u, b_f, CA, HB, 0, (float*)d_out);
    gu_level_k<<<dim3(512, 16), 128, 0, stream>>>(8192, HB, WPACK, GU);

    float* cc = CA;   // children c
    float* cp = CB;   // parent c
    for (int l = 12; l >= 0; --l) {
        int m = 1 << l;
        int off = m - 1;
        node_level_k<<<dim3((m + 15) / 16, 16), 128, 0, stream>>>(
            1, m, nullptr, nullptr, op_emb, op_ids, off,
            WPACK, cc, GU, INITV, c_init,
            b_i, b_o, b_u, b_f, cp, HB, (l == 0) ? 1 : 0, (float*)d_out);
        if (l > 0)
            gu_level_k<<<dim3((m + 15) / 16, 16), 128, 0, stream>>>(m, HB, WPACK, GU);
        float* tmp = cc; cc = cp; cp = tmp;
    }
}
